// ChunkModule_37469294690854
// MI455X (gfx1250) — matrix operand order repaired and checked by reading, not compiled, not run on hardware
//
#include <hip/hip_runtime.h>
#include <math.h>

// ---------------- problem constants (from reference) ----------------
#define B_    2
#define L_    2048
#define C_    256
#define BL_   (B_ * L_)
#define DI_   512          // D_INNER
#define DS_   16           // D_STATE
#define DTR_  16           // DT_RANK
#define DBCW_ 48           // DT_RANK + 2*D_STATE
#define H_    8
#define HD_   32

typedef __attribute__((ext_vector_type(16))) _Float16 v16h;
typedef __attribute__((ext_vector_type(8)))  float    v8f;

enum { ACT_NONE = 0, ACT_GELU = 1, ACT_SILU = 2, ACT_SOFTPLUS = 3 };

__device__ __forceinline__ float act_f(float y, int act) {
  switch (act) {
    case ACT_GELU:     return 0.5f * y * (1.f + erff(y * 0.7071067811865475f));
    case ACT_SILU:     return y / (1.f + __expf(-y));
    case ACT_SOFTPLUS: return (y > 20.f) ? y : log1pf(__expf(y));
    default:           return y;
  }
}

// =====================================================================
// Generic WMMA GEMM / conv-as-shifted-GEMM kernel.
//   C[m, n] = act( alpha * sum_t sum_ci A[b, l + t - pad, ci] * W(...)
//                  + bias[n] ) + res[m, n]
// A is (nb*L, CI) row-major with row stride lda (channels-last).
// blayout==0: W is (CO, CI, KT) row-major (weights / K-as-rows).
// blayout==1: W is (CI, CO) (plain B matrix, KT must be 1).
//
// Block: 128 threads = 4 wave32s in a 2x2 arrangement; each wave owns a
// 32x32 register tile = 2x2 WMMA 16x16 tiles (4 v_wmma per K-step with
// A-fragment reuse across N and B-fragment reuse across M).
// A tile is stored in LDS in natural K order (As[m][k]); A fragment uses
// k = lh*16+i, the same order as B, so each
// fragment is one contiguous 32B v16h load. B fragment (k = lh*16+i) is
// naturally contiguous.
// =====================================================================
__global__ __launch_bounds__(128)
void gemm_conv_wmma(const float* __restrict__ A, const float* __restrict__ W,
                    const float* __restrict__ bias, const float* __restrict__ res,
                    float* __restrict__ Cout,
                    int nb, int L, int CI, int CO, int lda, int ldc,
                    int KT, int pad, int blayout, int act, float alpha) {
  __shared__ __attribute__((aligned(64))) _Float16 As[64][32];
  __shared__ __attribute__((aligned(64))) _Float16 Bs[64][32];

  const int tid  = threadIdx.x;
  const int lane = tid & 31;
  const int wave = tid >> 5;          // 0..3
  const int wm   = wave >> 1;         // 0..1  (M sub-tile, 32 rows)
  const int wn   = wave & 1;          // 0..1  (N sub-tile, 32 cols)
  const int lh   = lane >> 4;         // lane half
  const int r    = lane & 15;
  const int m0   = blockIdx.y * 64;
  const int n0   = blockIdx.x * 64;
  const int Mtot = nb * L;

  v8f acc00 = {}, acc01 = {}, acc10 = {}, acc11 = {};

  for (int t = 0; t < KT; ++t) {
    const int shift = t - pad;
    for (int k0 = 0; k0 < CI; k0 += 32) {
      // ---- stage A tile (fp32 -> f16), As[m][k] natural order ----
      for (int i = tid; i < 64 * 32; i += 128) {
        int rr = i >> 5, cc = i & 31;
        int m = m0 + rr;
        float v = 0.f;
        if (m < Mtot) {
          int b = m / L;
          int l = m - b * L;
          int ls = l + shift;
          int ci = k0 + cc;
          if (ls >= 0 && ls < L && ci < CI)
            v = A[(size_t)(b * L + ls) * lda + ci];
        }
        As[rr][cc] = (_Float16)v;
      }
      // ---- stage B tile (fp32 -> f16), Bs[n][k] natural order ----
      for (int i = tid; i < 64 * 32; i += 128) {
        int nn = i >> 5, cc = i & 31;
        int n = n0 + nn;
        int ci = k0 + cc;
        float v = 0.f;
        if (n < CO && ci < CI) {
          if (blayout == 0) v = W[((size_t)n * CI + ci) * KT + t];
          else              v = W[(size_t)ci * CO + n];
        }
        Bs[nn][cc] = (_Float16)v;
      }
      // prefetch next A K-slab while we compute (gfx1250 global_prefetch_b8)
      if (k0 + 32 < CI)
        __builtin_prefetch(&A[(size_t)(m0 + (tid >> 1)) * lda + k0 + 32 + (tid & 1) * 16], 0, 1);
      __syncthreads();

      // ---- vectorized fragment loads (32B contiguous per lane) ----
      v16h a0 = *(const v16h*)&As[wm * 32 + r][lh * 16];
      v16h a1 = *(const v16h*)&As[wm * 32 + 16 + r][lh * 16];
      v16h b0 = *(const v16h*)&Bs[wn * 32 + r][lh * 16];
      v16h b1 = *(const v16h*)&Bs[wn * 32 + 16 + r][lh * 16];

      acc00 = __builtin_amdgcn_wmma_f32_16x16x32_f16(false, a0, false, b0, (short)0, acc00, false, false);
      acc01 = __builtin_amdgcn_wmma_f32_16x16x32_f16(false, a0, false, b1, (short)0, acc01, false, false);
      acc10 = __builtin_amdgcn_wmma_f32_16x16x32_f16(false, a1, false, b0, (short)0, acc10, false, false);
      acc11 = __builtin_amdgcn_wmma_f32_16x16x32_f16(false, a1, false, b1, (short)0, acc11, false, false);
      __syncthreads();
    }
  }

  // ---- store C (VGPR v: row = lh*8+v, col = lane&15) ----
  const v8f* accs[4] = { &acc00, &acc01, &acc10, &acc11 };
#pragma unroll
  for (int mi = 0; mi < 2; ++mi) {
#pragma unroll
    for (int ni = 0; ni < 2; ++ni) {
      const v8f& a = *accs[mi * 2 + ni];
#pragma unroll
      for (int v = 0; v < 8; ++v) {
        int grow = m0 + wm * 32 + mi * 16 + lh * 8 + v;
        int gcol = n0 + wn * 32 + ni * 16 + r;
        if (grow < Mtot && gcol < CO) {
          float y = a[v] * alpha;
          if (bias) y += bias[gcol];
          y = act_f(y, act);
          if (res) y += res[(size_t)grow * ldc + gcol];
          Cout[(size_t)grow * ldc + gcol] = y;
        }
      }
    }
  }
}

// =====================================================================
// Elementwise / reduction helper kernels
// =====================================================================
__global__ void layernorm_kernel(const float* __restrict__ x, const float* __restrict__ addv,
                                 const float* __restrict__ g, const float* __restrict__ b,
                                 float* __restrict__ out, int N) {
  __shared__ float sb[256];
  const int row = blockIdx.x;
  const float* xr = x + (size_t)row * N;
  const float* ar = addv ? addv + (size_t)row * N : nullptr;
  float s = 0.f;
  for (int i = threadIdx.x; i < N; i += 256) { float v = xr[i] + (ar ? ar[i] : 0.f); s += v; }
  sb[threadIdx.x] = s; __syncthreads();
  for (int k = 128; k > 0; k >>= 1) { if (threadIdx.x < k) sb[threadIdx.x] += sb[threadIdx.x + k]; __syncthreads(); }
  float mean = sb[0] / N; __syncthreads();
  float vs = 0.f;
  for (int i = threadIdx.x; i < N; i += 256) { float v = xr[i] + (ar ? ar[i] : 0.f) - mean; vs += v * v; }
  sb[threadIdx.x] = vs; __syncthreads();
  for (int k = 128; k > 0; k >>= 1) { if (threadIdx.x < k) sb[threadIdx.x] += sb[threadIdx.x + k]; __syncthreads(); }
  float inv = rsqrtf(sb[0] / N + 1e-5f);
  for (int i = threadIdx.x; i < N; i += 256) {
    float v = xr[i] + (ar ? ar[i] : 0.f);
    out[(size_t)row * N + i] = (v - mean) * inv * g[i] + b[i];
  }
}

__global__ void softmax_kernel(float* __restrict__ S, int N) {
  __shared__ float sb[256];
  const int row = blockIdx.x;
  float* r = S + (size_t)row * N;
  float m = -1e30f;
  for (int i = threadIdx.x; i < N; i += 256) m = fmaxf(m, r[i]);
  sb[threadIdx.x] = m; __syncthreads();
  for (int k = 128; k > 0; k >>= 1) { if (threadIdx.x < k) sb[threadIdx.x] = fmaxf(sb[threadIdx.x], sb[threadIdx.x + k]); __syncthreads(); }
  m = sb[0]; __syncthreads();
  float sum = 0.f;
  for (int i = threadIdx.x; i < N; i += 256) { float e = __expf(r[i] - m); r[i] = e; sum += e; }
  sb[threadIdx.x] = sum; __syncthreads();
  for (int k = 128; k > 0; k >>= 1) { if (threadIdx.x < k) sb[threadIdx.x] += sb[threadIdx.x + k]; __syncthreads(); }
  float inv = 1.f / sb[0];
  for (int i = threadIdx.x; i < N; i += 256) r[i] *= inv;
}

__global__ void bn_stats_kernel(const float* __restrict__ x, float* __restrict__ stats,
                                int M, int C) {
  __shared__ float s0[256], s1[256];
  const int c = blockIdx.x;
  float s = 0.f, sq = 0.f;
  for (int m = threadIdx.x; m < M; m += 256) {
    float v = x[(size_t)m * C + c];
    s += v; sq += v * v;
  }
  s0[threadIdx.x] = s; s1[threadIdx.x] = sq; __syncthreads();
  for (int k = 128; k > 0; k >>= 1) {
    if (threadIdx.x < k) { s0[threadIdx.x] += s0[threadIdx.x + k]; s1[threadIdx.x] += s1[threadIdx.x + k]; }
    __syncthreads();
  }
  if (threadIdx.x == 0) {
    float mean = s0[0] / M;
    stats[c] = mean;
    stats[C + c] = s1[0] / M - mean * mean;
  }
}

__global__ void bn_apply_kernel(const float* __restrict__ x, const float* __restrict__ stats,
                                const float* __restrict__ g, const float* __restrict__ b,
                                float* __restrict__ out, int M, int C, int gelu_after) {
  size_t idx = (size_t)blockIdx.x * blockDim.x + threadIdx.x;
  if (idx >= (size_t)M * C) return;
  int c = (int)(idx % C);
  float y = (x[idx] - stats[c]) * rsqrtf(stats[C + c] + 1e-5f) * g[c] + b[c];
  if (gelu_after) y = act_f(y, ACT_GELU);
  out[idx] = y;
}

// depthwise causal conv (k=4) over xz[:, :DI] + silu  ->  xs (B,L,DI)
__global__ void dwconv_silu_kernel(const float* __restrict__ xz, const float* __restrict__ cw,
                                   const float* __restrict__ cb, float* __restrict__ xs) {
  size_t idx = (size_t)blockIdx.x * blockDim.x + threadIdx.x;
  if (idx >= (size_t)BL_ * DI_) return;
  int d = (int)(idx % DI_);
  int l = (int)((idx / DI_) % L_);
  int b = (int)(idx / ((size_t)DI_ * L_));
  float acc = cb[d];
#pragma unroll
  for (int t = 0; t < 4; ++t) {
    int ls = l - 3 + t;
    if (ls >= 0)
      acc += cw[d * 4 + t] * xz[((size_t)(b * L_ + ls)) * (2 * DI_) + d];
  }
  xs[idx] = act_f(acc, ACT_SILU);
}

// selective-scan: one thread per (b, d); 16-state recurrence in registers
__global__ void ssm_scan_kernel(const float* __restrict__ dt, const float* __restrict__ xs,
                                const float* __restrict__ dbc, const float* __restrict__ A_log,
                                float* __restrict__ y) {
  int t = blockIdx.x * blockDim.x + threadIdx.x;
  if (t >= B_ * DI_) return;
  int d = t % DI_;
  int b = t / DI_;
  float A[DS_], h[DS_];
#pragma unroll
  for (int s = 0; s < DS_; ++s) { A[s] = -__expf(A_log[d * DS_ + s]); h[s] = 0.f; }
  for (int l = 0; l < L_; ++l) {
    size_t row = (size_t)b * L_ + l;
    float dtv = dt[row * DI_ + d];
    float xv  = xs[row * DI_ + d];
    float dx  = dtv * xv;
    const float* Br = dbc + row * DBCW_ + DTR_;
    const float* Cr = dbc + row * DBCW_ + DTR_ + DS_;
    float acc = 0.f;
#pragma unroll
    for (int s = 0; s < DS_; ++s) {
      h[s] = __expf(dtv * A[s]) * h[s] + dx * Br[s];
      acc += h[s] * Cr[s];
    }
    y[row * DI_ + d] = acc;
  }
}

// y = (yscan + xs * D[d]) * silu(z),  z = xz[:, DI + d]
__global__ void ymix_kernel(const float* __restrict__ yscan, const float* __restrict__ xs,
                            const float* __restrict__ Dp, const float* __restrict__ xz,
                            float* __restrict__ y) {
  size_t idx = (size_t)blockIdx.x * blockDim.x + threadIdx.x;
  if (idx >= (size_t)BL_ * DI_) return;
  int d = (int)(idx % DI_);
  size_t m = idx / DI_;
  float z = xz[m * (2 * DI_) + DI_ + d];
  y[idx] = (yscan[idx] + xs[idx] * Dp[d]) * act_f(z, ACT_SILU);
}

__global__ void flip_kernel(const float* __restrict__ in, float* __restrict__ out, int C) {
  size_t idx = (size_t)blockIdx.x * blockDim.x + threadIdx.x;
  if (idx >= (size_t)BL_ * C) return;
  int c = (int)(idx % C);
  size_t m = idx / C;
  int l = (int)(m % L_);
  int b = (int)(m / L_);
  out[((size_t)(b * L_ + (L_ - 1 - l))) * C + c] = in[idx];
}

__global__ void concat_kernel(const float* __restrict__ a, const float* __restrict__ b,
                              float* __restrict__ out, int C) {
  size_t idx = (size_t)blockIdx.x * blockDim.x + threadIdx.x;
  if (idx >= (size_t)BL_ * 2 * C) return;
  int j = (int)(idx % (2 * C));
  size_t m = idx / (2 * C);
  out[idx] = (j < C) ? a[m * C + j] : b[m * C + (j - C)];
}

__global__ void add_kernel(const float* __restrict__ a, const float* __restrict__ b,
                           float* __restrict__ out, size_t n) {
  size_t idx = (size_t)blockIdx.x * blockDim.x + threadIdx.x;
  if (idx < n) out[idx] = a[idx] + b[idx];
}

// (B,L,C) -> (B,H,L,HD)
__global__ void split_heads_kernel(const float* __restrict__ x, float* __restrict__ xh) {
  size_t idx = (size_t)blockIdx.x * blockDim.x + threadIdx.x;
  if (idx >= (size_t)BL_ * C_) return;
  int c = (int)(idx % C_);
  size_t m = idx / C_;
  int l = (int)(m % L_);
  int b = (int)(m / L_);
  int h = c / HD_, dd = c % HD_;
  xh[(((size_t)(b * H_ + h)) * L_ + l) * HD_ + dd] = x[idx];
}

// =====================================================================
// Host-side launch helpers
// =====================================================================
static inline dim3 g1(size_t n) { return dim3((unsigned)((n + 255) / 256)); }

static void run_gemm(hipStream_t s, const float* A, const float* W, const float* bias,
                     const float* res, float* C, int nb, int L, int CI, int CO,
                     int lda, int ldc, int KT, int pad, int blayout, int act, float alpha) {
  dim3 grid((CO + 63) / 64, (nb * L + 63) / 64);
  gemm_conv_wmma<<<grid, 128, 0, s>>>(A, W, bias, res, C, nb, L, CI, CO, lda, ldc,
                                      KT, pad, blayout, act, alpha);
}

// full-channel conv, channels-last, fused GELU
static void run_conv(hipStream_t s, const float* x, const float* w, const float* b,
                     float* out, int k, int pad) {
  run_gemm(s, x, w, b, nullptr, out, B_, L_, C_, C_, C_, C_, k, pad, 0, ACT_GELU, 1.f);
}

struct MambaP {
  const float *in_proj_w, *conv_w, *conv_b, *x_proj_w, *dt_proj_w, *dt_proj_b,
              *A_log, *D, *out_proj_w;
};

struct MambaWS { float *xz, *xs, *dbc, *dtb, *yscan, *ybuf; };

static void run_mamba(hipStream_t s, const float* in, const MambaP& p, float* out,
                      const MambaWS& w) {
  // in_proj: (BL,256) -> (BL,1024)
  run_gemm(s, in, p.in_proj_w, nullptr, nullptr, w.xz, 1, BL_, C_, 2 * DI_, C_, 2 * DI_,
           1, 0, 0, ACT_NONE, 1.f);
  dwconv_silu_kernel<<<g1((size_t)BL_ * DI_), 256, 0, s>>>(w.xz, p.conv_w, p.conv_b, w.xs);
  // x_proj: (BL,512) -> (BL,48)
  run_gemm(s, w.xs, p.x_proj_w, nullptr, nullptr, w.dbc, 1, BL_, DI_, DBCW_, DI_, DBCW_,
           1, 0, 0, ACT_NONE, 1.f);
  // dt_proj + softplus: (BL,16) -> (BL,512)
  run_gemm(s, w.dbc, p.dt_proj_w, p.dt_proj_b, nullptr, w.dtb, 1, BL_, DTR_, DI_, DBCW_,
           DI_, 1, 0, 0, ACT_SOFTPLUS, 1.f);
  ssm_scan_kernel<<<g1(B_ * DI_), 256, 0, s>>>(w.dtb, w.xs, w.dbc, p.A_log, w.yscan);
  ymix_kernel<<<g1((size_t)BL_ * DI_), 256, 0, s>>>(w.yscan, w.xs, p.D, w.xz, w.ybuf);
  // out_proj: (BL,512) -> (BL,256)
  run_gemm(s, w.ybuf, p.out_proj_w, nullptr, nullptr, out, 1, BL_, DI_, C_, DI_, C_,
           1, 0, 0, ACT_NONE, 1.f);
}

// conv_block; P points at the 18 cb leaves
static void run_conv_block(hipStream_t s, const float* x, const float* const* P,
                           float* out, float* t1, float* t2, float* t3, float* t4,
                           float* stats) {
  const float *c1w = P[0],  *c1b = P[1],  *c2w = P[2], *c2b = P[3];
  const float *c3w = P[4],  *c3b = P[5],  *l1w = P[6], *l1b = P[7];
  const float *l2w = P[8],  *l2b = P[9],  *fw  = P[10], *fb = P[11];
  const float *rw  = P[12], *rb  = P[13];
  const float *bn1g = P[14], *bn1b = P[15], *bn2g = P[16], *bn2b = P[17];

  run_conv(s, x,  c1w, c1b, t1, 3, 1);    // xa
  run_conv(s, t1, c2w, c2b, t2, 5, 2);
  run_conv(s, t2, c3w, c3b, t3, 7, 3);    // xb
  run_conv(s, t1, l1w, l1b, t2, 7, 3);
  run_conv(s, t2, l2w, l2b, t4, 15, 7);   // xc
  add_kernel<<<g1((size_t)BL_ * C_), 256, 0, s>>>(t3, t4, t2, (size_t)BL_ * C_);
  run_conv(s, t2, fw, fb, t3, 3, 1);
  bn_stats_kernel<<<C_, 256, 0, s>>>(t3, stats, BL_, C_);
  bn_apply_kernel<<<g1((size_t)BL_ * C_), 256, 0, s>>>(t3, stats, bn1g, bn1b, t4, BL_, C_, 0);
  run_conv(s, t4, rw, rb, t3, 1, 0);
  bn_stats_kernel<<<C_, 256, 0, s>>>(t3, stats, BL_, C_);
  bn_apply_kernel<<<g1((size_t)BL_ * C_), 256, 0, s>>>(t3, stats, bn2g, bn2b, out, BL_, C_, 1);
}

// =====================================================================
// kernel_launch
// =====================================================================
extern "C" void kernel_launch(void* const* d_in, const int* in_sizes, int n_in,
                              void* d_out, int out_size, void* d_ws, size_t ws_size,
                              hipStream_t stream) {
  (void)in_sizes; (void)out_size; (void)ws_size;
  const float* in[128];
  for (int i = 0; i < n_in && i < 128; ++i) in[i] = (const float*)d_in[i];

  // ---- parameter table (setup_inputs dict insertion order) ----
  const float* x = in[0];
  const float *bm_n1g = in[1], *bm_n1b = in[2], *bm_n2g = in[3], *bm_n2b = in[4];
  MambaP mf = { in[5],  in[6],  in[7],  in[8],  in[9],  in[10], in[11], in[12], in[13] };
  MambaP mb = { in[14], in[15], in[16], in[17], in[18], in[19], in[20], in[21], in[22] };
  const float *bm_f1w = in[23], *bm_f1b = in[24], *bm_f2w = in[25], *bm_f2b = in[26];
  const float* const* cb1 = &in[27];
  const float* const* cb2 = &in[45];
  const float* const* cb3 = &in[63];
  const float *ca_n1g = in[81], *ca_n1b = in[82], *ca_n2g = in[83], *ca_n2b = in[84];
  const float *ca_n3g = in[85], *ca_n3b = in[86];
  const float *ca_inw = in[87], *ca_inb = in[88], *ca_ow = in[89], *ca_ob = in[90];
  const float *ca_f1w = in[91], *ca_f1b = in[92], *ca_f2w = in[93], *ca_f2b = in[94];
  const float *dense_w = in[95], *dense_b = in[96];

  // ---- bump allocator over d_ws ----
  char* wp = (char*)d_ws;
  auto alloc = [&](size_t nfloats) -> float* {
    float* r = (float*)wp;
    wp += ((nfloats * sizeof(float) + 255) / 256) * 256;
    return r;
  };
  const size_t M1 = (size_t)BL_ * C_;      // 1M floats
  float* xn   = alloc(M1);
  float* xnf  = alloc(M1);
  MambaWS mw;
  mw.xz    = alloc((size_t)BL_ * 2 * DI_);
  mw.xs    = alloc((size_t)BL_ * DI_);
  mw.dbc   = alloc((size_t)BL_ * DBCW_);
  mw.dtb   = alloc((size_t)BL_ * DI_);
  mw.yscan = alloc((size_t)BL_ * DI_);
  mw.ybuf  = alloc((size_t)BL_ * DI_);
  float* mfwd = alloc(M1);
  float* mtmp = alloc(M1);
  float* mbwd = alloc(M1);
  float* bi   = alloc((size_t)BL_ * 2 * C_);
  float* ffh  = alloc((size_t)BL_ * 4 * C_);
  float* fout = alloc(M1);
  float* xa0  = alloc(M1);
  float* cv1  = alloc(M1);
  float* xskip= alloc(M1);
  float* xd   = alloc(M1);
  float* cv2  = alloc(M1);
  float* caout= alloc(M1);
  float* cv3  = alloc(M1);
  float* t1 = alloc(M1); float* t2 = alloc(M1);
  float* t3 = alloc(M1); float* t4 = alloc(M1);
  float* stats = alloc(2 * C_);
  float* qin  = alloc(M1);
  float* kvin = alloc(M1);
  float* qb = alloc(M1); float* kb = alloc(M1); float* vb = alloc(M1);
  float* qs = alloc(M1); float* ks = alloc(M1); float* vs = alloc(M1);
  float* Sb = alloc((size_t)L_ * L_);
  float* ob = alloc(M1);
  float* ab = alloc(M1);
  float* a2 = alloc(M1);
  float* fb1 = alloc((size_t)BL_ * (C_ / 2));

  // ================= bimamba block =================
  layernorm_kernel<<<BL_, 256, 0, stream>>>(x, nullptr, bm_n1g, bm_n1b, xn, C_);
  flip_kernel<<<g1(M1), 256, 0, stream>>>(xn, xnf, C_);
  run_mamba(stream, xn, mf, mfwd, mw);
  run_mamba(stream, xnf, mb, mtmp, mw);
  flip_kernel<<<g1(M1), 256, 0, stream>>>(mtmp, mbwd, C_);
  concat_kernel<<<g1((size_t)BL_ * 2 * C_), 256, 0, stream>>>(mfwd, mbwd, bi, C_);
  run_gemm(stream, bi, bm_f1w, bm_f1b, nullptr, ffh, 1, BL_, 2 * C_, 4 * C_,
           2 * C_, 4 * C_, 1, 0, 0, ACT_GELU, 1.f);
  run_gemm(stream, ffh, bm_f2w, bm_f2b, nullptr, fout, 1, BL_, 4 * C_, C_,
           4 * C_, C_, 1, 0, 0, ACT_GELU, 1.f);
  layernorm_kernel<<<BL_, 256, 0, stream>>>(x, fout, bm_n2g, bm_n2b, xa0, C_);

  // ================= conv stack =================
  run_conv_block(stream, xa0, cb1, cv1, t1, t2, t3, t4, stats);
  run_conv_block(stream, cv1, cb2, xskip, t1, t2, t3, t4, stats);
  run_gemm(stream, cv1, dense_w, dense_b, nullptr, xd, 1, BL_, C_, C_, C_, C_,
           1, 0, 0, ACT_GELU, 1.f);
  run_conv_block(stream, xd, cb2, cv2, t1, t2, t3, t4, stats);

  // ================= cross attention =================
  layernorm_kernel<<<BL_, 256, 0, stream>>>(cv2, nullptr, ca_n1g, ca_n1b, qin, C_);
  layernorm_kernel<<<BL_, 256, 0, stream>>>(xa0, nullptr, ca_n2g, ca_n2b, kvin, C_);
  run_gemm(stream, qin,  ca_inw,              ca_inb,          nullptr, qb, 1, BL_, C_, C_,
           C_, C_, 1, 0, 0, ACT_NONE, 1.f);
  run_gemm(stream, kvin, ca_inw + (size_t)C_ * C_,     ca_inb + C_,     nullptr, kb, 1, BL_, C_, C_,
           C_, C_, 1, 0, 0, ACT_NONE, 1.f);
  run_gemm(stream, kvin, ca_inw + (size_t)2 * C_ * C_, ca_inb + 2 * C_, nullptr, vb, 1, BL_, C_, C_,
           C_, C_, 1, 0, 0, ACT_NONE, 1.f);
  split_heads_kernel<<<g1(M1), 256, 0, stream>>>(qb, qs);
  split_heads_kernel<<<g1(M1), 256, 0, stream>>>(kb, ks);
  split_heads_kernel<<<g1(M1), 256, 0, stream>>>(vb, vs);
  const float scale = 1.f / sqrtf((float)HD_);
  for (int b = 0; b < B_; ++b) {
    for (int h = 0; h < H_; ++h) {
      size_t base = ((size_t)(b * H_ + h)) * L_ * HD_;
      // S = scale * Q @ K^T   (K rows are contiguous -> layout 0)
      run_gemm(stream, qs + base, ks + base, nullptr, nullptr, Sb, 1, L_, HD_, L_,
               HD_, L_, 1, 0, 0, ACT_NONE, scale);
      softmax_kernel<<<L_, 256, 0, stream>>>(Sb, L_);
      // O = P @ V   (V is (K,N) -> layout 1); write head slice of ob
      run_gemm(stream, Sb, vs + base, nullptr, nullptr,
               ob + (size_t)b * L_ * C_ + h * HD_, 1, L_, L_, HD_, L_, C_,
               1, 0, 1, ACT_NONE, 1.f);
    }
  }
  run_gemm(stream, ob, ca_ow, ca_ob, qin, ab, 1, BL_, C_, C_, C_, C_,
           1, 0, 0, ACT_NONE, 1.f);                       // + q_in residual
  layernorm_kernel<<<BL_, 256, 0, stream>>>(ab, nullptr, ca_n3g, ca_n3b, a2, C_);
  run_gemm(stream, a2, ca_f1w, ca_f1b, nullptr, fb1, 1, BL_, C_, C_ / 2, C_, C_ / 2,
           1, 0, 0, ACT_GELU, 1.f);
  run_gemm(stream, fb1, ca_f2w, ca_f2b, a2, caout, 1, BL_, C_ / 2, C_, C_ / 2, C_,
           1, 0, 0, ACT_GELU, 1.f);                       // + a residual

  run_conv_block(stream, caout, cb3, cv3, t1, t2, t3, t4, stats);

  // ================= output: concat(xa_skip, xa) =================
  concat_kernel<<<g1((size_t)BL_ * 2 * C_), 256, 0, stream>>>(xskip, cv3, (float*)d_out, C_);
}
